// BMM_S8T_S8N_F32T_72292889526615
// MI455X (gfx1250) — 8-bit matrix operand order repaired and checked by reading, compiled, not run on hardware
//
#include <hip/hip_runtime.h>
#include <stdint.h>


typedef int v8i __attribute__((ext_vector_type(8)));

static constexpr int BATCH = 64;
static constexpr int M = 2048;
static constexpr int N = 2048;
static constexpr int K = 64;
static constexpr int MTILE = 128;
static constexpr int NTILE = 128;

__global__ __launch_bounds__(256)
void bmm_s8s8_f32_wmma(const int8_t* __restrict__ A,
                       const int8_t* __restrict__ Bm,
                       const float* __restrict__ alpha_p,
                       float* __restrict__ out)
{
    const int lane = threadIdx.x & 31;
    const int wave = threadIdx.x >> 5;
    const int g    = lane >> 4;
    const int ln   = lane & 15;

    const int batch = blockIdx.z;
    const int m0    = blockIdx.y * MTILE + wave * 16;
    const int n0    = blockIdx.x * NTILE;

    const float alpha = *alpha_p;

    const int8_t* arow = A + ((size_t)batch * M + (m0 + ln)) * K + g * 16;
    uint2 a01 = *(const uint2*)(arow +  0);
    uint2 a23 = *(const uint2*)(arow +  8);
    uint2 a45 = *(const uint2*)(arow + 32);
    uint2 a67 = *(const uint2*)(arow + 40);
    v8i afrag = { (int)a01.x, (int)a01.y, (int)a23.x, (int)a23.y,
                  (int)a45.x, (int)a45.y, (int)a67.x, (int)a67.y };

    const int8_t* bbase = Bm + ((size_t)batch * N + (n0 + ln)) * K + g * 16;

    float* obase = out + ((size_t)batch * M + m0 + 8 * g) * (size_t)N + n0 + ln;

#pragma unroll
    for (int nt = 0; nt < NTILE / 16; ++nt) {
        const int8_t* bp = bbase + (size_t)nt * 16 * K;
        uint4 blo = *(const uint4*)(bp +  0);
        uint4 bhi = *(const uint4*)(bp + 32);
        v8i bfrag = { (int)blo.x, (int)blo.y, (int)blo.z, (int)blo.w,
                      (int)bhi.x, (int)bhi.y, (int)bhi.z, (int)bhi.w };

        v8i acc = {0, 0, 0, 0, 0, 0, 0, 0};
        acc = __builtin_amdgcn_wmma_i32_16x16x64_iu8(
            true, afrag, true, bfrag, acc, false, false);

        float* op = obase + nt * 16;
#pragma unroll
        for (int r = 0; r < 8; ++r) {
            __builtin_nontemporal_store(alpha * (float)acc[r],
                                        op + (size_t)r * N);
        }
    }
}

extern "C" void kernel_launch(void* const* d_in, const int* in_sizes, int n_in,
                              void* d_out, int out_size, void* d_ws, size_t ws_size,
                              hipStream_t stream) {
    const int8_t* A     = (const int8_t*)d_in[0];
    const int8_t* Bm    = (const int8_t*)d_in[1];
    const float*  alpha = (const float*)d_in[2];
    float*        out   = (float*)d_out;

    dim3 grid(N / NTILE, M / MTILE, BATCH);
    dim3 block(256);
    bmm_s8s8_f32_wmma<<<grid, block, 0, stream>>>(A, Bm, alpha, out);
}
